// MinCutExplainerGNN_44770739093935
// MI455X (gfx1250) — matrix operand order repaired and checked by reading, not compiled, not run on hardware
//
#include <hip/hip_runtime.h>

// ---------------- problem sizes ----------------
#define NN 100000
#define EE 3200000
#define DD 512
#define HH 512
#define OO 128
#define KK 30
#define KP 32
#define EPSF 1e-9f

typedef __attribute__((ext_vector_type(16))) __bf16 v16bf;
typedef __attribute__((ext_vector_type(16))) short  v16s;
typedef __attribute__((ext_vector_type(8)))  short  v8s;
typedef __attribute__((ext_vector_type(8)))  float  v8f;
typedef __attribute__((ext_vector_type(4)))  unsigned int u32x4;
typedef __attribute__((ext_vector_type(8)))  int    i32x8;
typedef __attribute__((ext_vector_type(4)))  int    i32x4;

#if defined(__has_builtin)
#if __has_builtin(__builtin_amdgcn_tensor_load_to_lds) && __has_builtin(__builtin_amdgcn_s_wait_tensorcnt)
#define HAVE_TDM 1
#endif
#endif

// ---------------- workspace layout (float offsets) ----------------
#define SP_OFF   0u         // S padded [N][32]
#define DEG_OFF  6400000u   // deg [N]
#define ZG_OFF   6500096u   // Z accum [32][512]
#define SS_OFF   6516480u   // S^T S [32][32]
#define ADJ_OFF  6517504u   // S^T A S [32][32] (edge outer products)
#define VOL_OFF  6518528u   // scalars
#define EA_OFF   6520000u   // explainer a [30][512]
#define EB_OFF   6535360u   // explainer b [30][512]
#define EM_OFF   6550720u   // m matrix [32][32]
#define EEMB_OFF 6551744u   // out_emb [30][512]
#define H1_OFF   6567104u   // h1 [30][512]
#define H2_OFF   6582464u   // h2 [30][512]
#define WPB_OFF  6600000u   // Wp^T bf16 [512 cols][512 d]  (131072 floats)
#define WAB_OFF  6731072u   // Wa^T bf16 [32 cols][512 d]   (8192 floats)
#define XBF_OFF  6739264u   // x bf16 [N][512]              (25.6M floats)

// ---------------- output layout (float offsets) ----------------
#define O_OUT 0
#define O_ML  3840
#define O_OL  3841
#define O_Z   3842
#define O_S   19202

// ---------------- helpers ----------------
__device__ __forceinline__ unsigned short f2bfbits(float f) {
  unsigned u = __builtin_bit_cast(unsigned, f);
  unsigned r = u + 0x7FFFu + ((u >> 16) & 1u);   // round-to-nearest-even
  return (unsigned short)(r >> 16);
}

// A fragment 16x32 bf16 from row-major LDS buf[row][k] (ISA 7.12.2):
// lane (m = lane&15, h = lane>>4) holds K = {8h..8h+7} and {16+8h..23+8h} -> 2 contiguous b128 loads
__device__ __forceinline__ v16bf load_a_frag(const unsigned short* buf, int row0, int k0, int stride) {
  int lane = threadIdx.x & 31;
  int m = lane & 15, half = lane >> 4;
  const unsigned short* base = buf + (row0 + m) * stride + k0 + 8 * half;
  v8s lo = *reinterpret_cast<const v8s*>(base);
  v8s hi = *reinterpret_cast<const v8s*>(base + 16);
  v16s s = __builtin_shufflevector(lo, hi, 0, 1, 2, 3, 4, 5, 6, 7, 8, 9, 10, 11, 12, 13, 14, 15);
  return __builtin_bit_cast(v16bf, s);
}

// B fragment 32x16 bf16 from K-major LDS buf[col][k]:
// lane (n = lane&15, h = lane>>4) holds K = {8h..8h+7} and {16+8h..23+8h} -> 2 contiguous b128 loads
__device__ __forceinline__ v16bf load_b_frag_t(const unsigned short* buf, int col0, int k0, int stride) {
  int lane = threadIdx.x & 31;
  int n = lane & 15, half = lane >> 4;
  const unsigned short* base = buf + (col0 + n) * stride + k0 + 8 * half;
  v8s lo = *reinterpret_cast<const v8s*>(base);
  v8s hi = *reinterpret_cast<const v8s*>(base + 16);
  v16s s = __builtin_shufflevector(lo, hi, 0, 1, 2, 3, 4, 5, 6, 7, 8, 9, 10, 11, 12, 13, 14, 15);
  return __builtin_bit_cast(v16bf, s);
}

// ---------------- kernel 0a: x -> bf16 (then L2-resident for both GEMM passes) ----------------
__global__ void k_convx(const float* __restrict__ x, unsigned short* __restrict__ xbf) {
  long i8 = (long)blockIdx.x * blockDim.x + threadIdx.x;   // 8 elements per thread
  if (i8 >= (long)NN * DD / 8) return;
  long base = i8 * 8;
  const float4* src = (const float4*)(x + base);
  float4 a = src[0], b = src[1];
  v8s o;
  o[0] = (short)f2bfbits(a.x); o[1] = (short)f2bfbits(a.y);
  o[2] = (short)f2bfbits(a.z); o[3] = (short)f2bfbits(a.w);
  o[4] = (short)f2bfbits(b.x); o[5] = (short)f2bfbits(b.y);
  o[6] = (short)f2bfbits(b.z); o[7] = (short)f2bfbits(b.w);
  *reinterpret_cast<v8s*>(xbf + base) = o;
}

// ---------------- kernel 0b: zero accumulators + transpose-convert weights ----------------
__global__ void k_prep(const float* __restrict__ Wp, const float* __restrict__ Wa,
                       float* __restrict__ deg, float* __restrict__ ZG,
                       float* __restrict__ SS, float* __restrict__ ADJ, float* __restrict__ VOL,
                       unsigned short* __restrict__ wpb, unsigned short* __restrict__ wab) {
  long idx = (long)blockIdx.x * blockDim.x + threadIdx.x;
  if (idx < NN) deg[idx] = 0.f;
  if (idx < KP * DD) ZG[idx] = 0.f;
  if (idx < KP * KP) { SS[idx] = 0.f; ADJ[idx] = 0.f; }
  if (idx < 16) VOL[idx] = 0.f;
  if (idx < (long)DD * DD) {                 // Wp^T: wpb[c][d] = Wp[d][c]
    int c = (int)(idx >> 9), d = (int)(idx & 511);
    wpb[idx] = f2bfbits(Wp[(long)d * DD + c]);
  }
  if (idx < KP * DD) {                       // Wa^T: wab[c][d] = Wa[d][c], pad c>=30 with 0
    int c = (int)(idx >> 9), d = (int)(idx & 511);
    wab[idx] = (c < KK) ? f2bfbits(Wa[d * KK + c]) : (unsigned short)0;
  }
}

// ---------------- kernel 1: S = softmax(x @ Wa + ba)  (WMMA bf16, persistent blocks) ---------
__global__ void k_s(const unsigned short* __restrict__ xbf, const unsigned short* __restrict__ wab,
                    const float* __restrict__ ba, float* __restrict__ Sp, float* __restrict__ outS) {
  __shared__ unsigned short xb[32 * 528];    // x tile, row-major [node][k]
  __shared__ unsigned short wbt[32 * 520];   // Wa^T, K-major [col][k] (resident all tiles)
  __shared__ float lg[32 * 33];              // logits
  int tid = threadIdx.x;

  for (int i = tid; i < 32 * 64; i += 128) {                 // Wa^T: loaded once per block
    int c = i >> 6, d8 = (i & 63) * 8;
    *reinterpret_cast<v8s*>(wbt + c * 520 + d8) =
        *reinterpret_cast<const v8s*>(wab + c * DD + d8);
  }

  int wv = tid >> 5, lane = tid & 31, half = lane >> 4, n = lane & 15;
  int rt = wv & 1, ct = wv >> 1;             // 4 waves: 2 row tiles x 2 col tiles
  const int NT = NN / 32;                    // 3125 node tiles
  for (int t = blockIdx.x; t < NT; t += gridDim.x) {
    long n0 = (long)t * 32;
    __syncthreads();                         // protects xb/lg from previous iteration
    for (int i = tid; i < 32 * 64; i += 128) {
      int r = i >> 6, c8 = (i & 63) * 8;
      *reinterpret_cast<v8s*>(xb + r * 528 + c8) =
          *reinterpret_cast<const v8s*>(xbf + (n0 + r) * DD + c8);
    }
    __syncthreads();

    v8f acc; for (int i = 0; i < 8; ++i) acc[i] = 0.f;
#pragma unroll 4
    for (int k0 = 0; k0 < 512; k0 += 32) {
      v16bf a = load_a_frag(xb, rt * 16, k0, 528);
      v16bf b = load_b_frag_t(wbt, ct * 16, k0, 520);
      acc = __builtin_amdgcn_wmma_f32_16x16x32_bf16(false, a, false, b, (short)0, acc, false, false);
    }
#pragma unroll
    for (int r = 0; r < 8; ++r)
      lg[(rt * 16 + r + 8 * half) * 33 + ct * 16 + n] = acc[r];
    __syncthreads();

    if (tid < 32) {                          // per-row softmax over K=30
      long node = n0 + tid;
      float mx = -1e30f;
      for (int k = 0; k < KK; ++k) mx = fmaxf(mx, lg[tid * 33 + k] + ba[k]);
      float sum = 0.f;
      for (int k = 0; k < KK; ++k) {
        float e = __expf(lg[tid * 33 + k] + ba[k] - mx);
        lg[tid * 33 + k] = e;
        sum += e;
      }
      float inv = 1.f / sum;
      for (int k = 0; k < KK; ++k) {
        float sv = lg[tid * 33 + k] * inv;
        Sp[node * KP + k] = sv;
        outS[node * KK + k] = sv;
      }
      Sp[node * KP + 30] = 0.f;
      Sp[node * KP + 31] = 0.f;
    }
  }
}

// ---------------- kernel 2: ADJ = sum_e S[row_e] (x) S[col_e] ; deg[row] += 1 ----------------
// Per-wave 32x32 accumulator in VGPRs (lane = column j); v_readlane broadcasts S[row][i].
__global__ void k_edges(const int* __restrict__ ei, const float* __restrict__ Sp,
                        float* __restrict__ ADJ, float* __restrict__ deg) {
  int tid = threadIdx.x, lane = tid & 31;
  int gw = blockIdx.x * (blockDim.x >> 5) + (tid >> 5);
  int nw = gridDim.x * (blockDim.x >> 5);
  float acc[32];
#pragma unroll
  for (int i = 0; i < 32; ++i) acc[i] = 0.f;

  const int NCH = EE / 32;                   // 100000 chunks of 32 edges
  for (int ch = gw; ch < NCH; ch += nw) {
    long e0 = (long)ch * 32;
    int rb = ei[e0 + lane];                  // coalesced index loads
    int cb = ei[EE + e0 + lane];
#pragma unroll 1
    for (int i = 0; i < 32; ++i) {
      int r = __shfl(rb, i, 32);
      int c = __shfl(cb, i, 32);
      float sc = Sp[(long)c * KP + lane];    // lane's column of S[col]
      float sr = Sp[(long)r * KP + lane];    // lane's column of S[row] (for readlane)
      if (lane == 0) atomicAdd(deg + r, 1.f);
#pragma unroll
      for (int ii = 0; ii < 32; ++ii)
        acc[ii] += __shfl(sr, ii, 32) * sc;  // rank-1 update
    }
  }
#pragma unroll
  for (int i = 0; i < 32; ++i)
    atomicAdd(&ADJ[i * KP + lane], acc[i]);
}

// ---------------- kernel 3: SS = S^T S ; vol = sum(deg * S) ----------------
__global__ void k_reduce(const float* __restrict__ Sp, const float* __restrict__ deg,
                         float* __restrict__ SS, float* __restrict__ VOL) {
  __shared__ float S8[8][33];
  int tid = threadIdx.x;
  int j = tid & 31, i0 = tid >> 5;           // thread owns (i0+8q, j) for q=0..3
  float ss[4] = {0, 0, 0, 0};
  float volp = 0.f;
  int nb = (NN + 7) / 8;
  for (int b = blockIdx.x; b < nb; b += gridDim.x) {
    __syncthreads();
    {
      int nib = tid >> 5, k = tid & 31;
      long nd = (long)b * 8 + nib;
      S8[nib][k] = (nd < NN) ? Sp[nd * KP + k] : 0.f;
    }
    __syncthreads();
#pragma unroll
    for (int nib = 0; nib < 8; ++nib) {
      float sj = S8[nib][j];
#pragma unroll
      for (int q = 0; q < 4; ++q) ss[q] += S8[nib][i0 + 8 * q] * sj;
    }
    if (tid < 8) {
      long nd = (long)b * 8 + tid;
      if (nd < NN) {
        float rs = 0.f;
        for (int k = 0; k < KP; ++k) rs += S8[tid][k];
        volp += rs * deg[nd];
      }
    }
  }
#pragma unroll
  for (int q = 0; q < 4; ++q) atomicAdd(&SS[(i0 + 8 * q) * KP + j], ss[q]);
  if (tid < 8) atomicAdd(VOL, volp);
}

// ---------------- kernel 4: losses ----------------
__global__ void k_loss(const float* __restrict__ SS, const float* __restrict__ ADJ,
                       const float* __restrict__ VOL, float* __restrict__ out_ml,
                       float* __restrict__ out_ol) {
  if (threadIdx.x == 0) {
    float cut = 0.f;
    for (int k = 0; k < KK; ++k) cut += ADJ[k * KP + k];
    out_ml[0] = -cut / (VOL[0] + EPSF);
    float o = 0.f;
    for (int i = 0; i < KK; ++i)
      for (int j2 = 0; j2 < KK; ++j2) {
        float d = SS[i * KP + j2] - ((i == j2) ? 1.f : 0.f);
        o += d * d;
      }
    out_ol[0] = sqrtf(o);
  }
}

// ---------------- kernel 5: fused Xp = x@Wp+bp ; Z += S^T @ Xp (double WMMA + TDM) ----------
__global__ void __launch_bounds__(256) k_xpz(const unsigned short* __restrict__ xbf,
                                             const unsigned short* __restrict__ wpb,
                                             const float* __restrict__ bp,
                                             const float* __restrict__ Sp,
                                             float* __restrict__ ZG) {
  __shared__ unsigned short wpl[64 * 520];   // Wp^T slice, K-major [col][d]
  __shared__ unsigned short xb[32 * 528];    // x tile, row-major [node][d] (528 = 512 + 16 pad)
  __shared__ unsigned short xpt[64 * 40];    // Xp^T tile, K-major [col][node]
  __shared__ unsigned short sbt[32 * 40];    // S^T tile, row-major [cluster][node]
  __shared__ float bps[64];
  int tid = threadIdx.x;
  int cg = blockIdx.x & 7;                   // column group: 64 cols of D
  int sp = blockIdx.x >> 3;                  // node split
  int nsp = gridDim.x >> 3;

  for (int i = tid; i < 64 * 64; i += 256) { // Wp^T slice: straight vector copy
    int c = i >> 6, d8 = (i & 63) * 8;
    *reinterpret_cast<v8s*>(wpl + c * 520 + d8) =
        *reinterpret_cast<const v8s*>(wpb + (long)(cg * 64 + c) * DD + d8);
  }
  if (tid < 64) bps[tid] = bp[cg * 64 + tid];

  int wv = tid >> 5, lane = tid & 31, half = lane >> 4, n = lane & 15;
  int rt = wv & 1, ct = wv >> 1;             // 8 waves: 2 row tiles x 4 col tiles
  v8f zacc; for (int i = 0; i < 8; ++i) zacc[i] = 0.f;

  const int NT = NN / 32;                    // 3125 node tiles
  for (int t = sp; t < NT; t += nsp) {
    long n0 = (long)t * 32;
    __syncthreads();                         // protect LDS from previous iteration

#if defined(HAVE_TDM)
    if (tid < 32) {                          // wave 0 issues a TDM 2D tile load:
      // 32 rows x 512 bf16, row stride 512 elems; LDS pad 8 dwords after every 256 dwords
      // (one 1024-B row) -> reproduces the 528-short padded row layout. ISA 8.3/8.4.
      unsigned long ga = (unsigned long)(const void*)xbf + (unsigned long)n0 * (DD * 2);
      unsigned lds = (unsigned)(unsigned long)(const void*)xb;
      u32x4 g0;
      g0[0] = 1u;                                            // count=1, user descriptor
      g0[1] = lds;                                           // lds_addr
      g0[2] = (unsigned)(ga & 0xFFFFFFFFu);                  // global_addr[31:0]
      g0[3] = (unsigned)((ga >> 32) & 0x1FFFFFFu) | (2u << 30); // global_addr[56:32] | type=2
      i32x8 g1;
      g1[0] = (1 << 16) | (1 << 20) | (7 << 22) | (7 << 25); // data_size=2B, pad_en, ivl=256dw, amt=8dw
      g1[1] = (int)((DD & 0xFFFFu) << 16);                   // tensor_dim0 = 512 (bits 79:48)
      g1[2] = (int)(((unsigned)NN & 0xFFFFu) << 16);         // tensor_dim1[15:0] (bits 95:80)
      g1[3] = (int)((NN >> 16) | ((unsigned)DD << 16));      // tensor_dim1 hi | tile_dim0=512
      g1[4] = 32;                                            // tile_dim1=32, tile_dim2=0
      g1[5] = DD;                                            // tensor_dim0_stride = 512
      g1[6] = 0; g1[7] = 0;
      i32x4 gz4 = {0, 0, 0, 0};
      i32x8 gz8 = {0, 0, 0, 0, 0, 0, 0, 0};
      // 6-arg toolchain form: (g0, g1, g2, g3, extra, cpol)
      __builtin_amdgcn_tensor_load_to_lds(g0, g1, gz4, gz4, gz8, 0);
    }
#else
    for (int i = tid; i < 32 * 64; i += 256) {
      int r = i >> 6, c8 = (i & 63) * 8;
      *reinterpret_cast<v8s*>(xb + r * 528 + c8) =
          *reinterpret_cast<const v8s*>(xbf + (n0 + r) * DD + c8);
    }
#endif
    for (int i = tid; i < 32 * 32; i += 256) {
      int cl = i >> 5, nd = i & 31;          // sbt[cluster][node] = S[node][cluster]
      sbt[cl * 40 + nd] = f2bfbits(Sp[(n0 + nd) * KP + cl]);
    }
#if defined(HAVE_TDM)
    __builtin_amdgcn_s_wait_tensorcnt(0);    // issuing wave drains TDM; no-op for others
#endif
    __syncthreads();                         // publish xb + sbt to all waves

    // Phase 1: Xp tile = x_tile @ Wp_slice
    v8f acc; for (int i = 0; i < 8; ++i) acc[i] = 0.f;
#pragma unroll 4
    for (int k0 = 0; k0 < 512; k0 += 32) {
      v16bf a = load_a_frag(xb, rt * 16, k0, 528);
      v16bf b = load_b_frag_t(wpl, ct * 16, k0, 520);
      acc = __builtin_amdgcn_wmma_f32_16x16x32_bf16(false, a, false, b, (short)0, acc, false, false);
    }
#pragma unroll
    for (int r = 0; r < 8; ++r) {            // write Xp^T (K-major for phase-2 B)
      int row = rt * 16 + r + 8 * half;      // node
      int col = ct * 16 + n;                 // feature col
      xpt[col * 40 + row] = f2bfbits(acc[r] + bps[col]);
    }
    __syncthreads();

    // Phase 2: Z += S^T (clusters x 32 nodes) @ Xp (32 nodes x cols)
    v16bf a2 = load_a_frag(sbt, rt * 16, 0, 40);
    v16bf b2 = load_b_frag_t(xpt, ct * 16, 0, 40);
    zacc = __builtin_amdgcn_wmma_f32_16x16x32_bf16(false, a2, false, b2, (short)0, zacc, false, false);
  }
#pragma unroll
  for (int r = 0; r < 8; ++r) {
    int cl = rt * 16 + r + 8 * half;
    int col = cg * 64 + ct * 16 + n;
    atomicAdd(&ZG[cl * 512 + col], zacc[r]);
  }
}

// ---------------- kernel 6: copy Z to output ----------------
__global__ void k_copyz(const float* __restrict__ ZG, float* __restrict__ outZ) {
  int i = blockIdx.x * blockDim.x + threadIdx.x;
  if (i < KK * DD) outZ[i] = ZG[i];          // ZG rows 0..29 are contiguous
}

// ---------------- explainer (K=30 dense; tiny) ----------------
__global__ void k_eab(const float* __restrict__ Xin, const float* __restrict__ W1,
                      float* __restrict__ A, float* __restrict__ B, int din, int hid) {
  int idx = blockIdx.x * blockDim.x + threadIdx.x;
  int total = KK * hid;
  if (idx >= 2 * total) return;
  int which = idx >= total;
  int t = which ? idx - total : idx;
  int i = t / hid, c = t % hid;
  const float* Wc = W1 + (which ? (long)din * hid : 0);
  float s = 0.f;
  for (int k = 0; k < din; ++k) s += Xin[(long)i * din + k] * Wc[(long)k * hid + c];
  (which ? B : A)[t] = s;
}

__global__ void k_elog(const float* __restrict__ A, const float* __restrict__ B,
                       const float* __restrict__ b1, const float* __restrict__ W2,
                       const float* __restrict__ b2, const float* __restrict__ ADJ,
                       float* __restrict__ M, int hid) {
  int wv = (blockIdx.x * blockDim.x + threadIdx.x) >> 5;
  int lane = threadIdx.x & 31;
  if (wv >= KK * KK) return;
  int i = wv / KK, j = wv % KK;
  float s = 0.f;
  for (int c = lane; c < hid; c += 32) {
    float h = A[i * hid + c] + B[j * hid + c] + b1[c];
    s += fmaxf(h, 0.f) * W2[c];
  }
#pragma unroll
  for (int off = 16; off > 0; off >>= 1) s += __shfl_xor(s, off, 32);
  if (lane == 0) {
    float logit = s + b2[0];
    float mask = (ADJ[i * KP + j] > 0.f) ? 1.f : 0.f;
    M[i * KP + j] = (1.f / (1.f + __expf(-logit))) * mask;
  }
}

__global__ void k_eemb(const float* __restrict__ Xin, const float* __restrict__ M,
                       const float* __restrict__ ADJ, float* __restrict__ EMB, int din) {
  int idx = blockIdx.x * blockDim.x + threadIdx.x;
  if (idx >= KK * din) return;
  int i = idx / din, c = idx % din;
  float msum = 0.f;
  int hn = 0;
  for (int j = 0; j < KK; ++j) {
    msum += M[i * KP + j];
    hn |= (ADJ[i * KP + j] > 0.f) ? 1 : 0;
  }
  float inv = 1.f / (msum + EPSF);
  float agg = 0.f;
  for (int j = 0; j < KK; ++j) agg += M[i * KP + j] * inv * Xin[(long)j * din + c];
  float xi = Xin[(long)i * din + c];
  EMB[idx] = hn ? (0.5f * xi + 0.5f * agg) : xi;
}

__global__ void k_elin(const float* __restrict__ Xin, const float* __restrict__ W,
                       const float* __restrict__ bias, float* __restrict__ Y,
                       int din, int dout, int dorelu) {
  int idx = blockIdx.x * blockDim.x + threadIdx.x;
  if (idx >= KK * dout) return;
  int i = idx / dout, c = idx % dout;
  float s = bias[c];
  for (int k = 0; k < din; ++k) s += Xin[(long)i * din + k] * W[(long)k * dout + c];
  Y[idx] = dorelu ? fmaxf(s, 0.f) : s;
}

// ---------------- host launch ----------------
extern "C" void kernel_launch(void* const* d_in, const int* in_sizes, int n_in,
                              void* d_out, int out_size, void* d_ws, size_t ws_size,
                              hipStream_t stream) {
  (void)in_sizes; (void)n_in; (void)out_size; (void)ws_size;
  const float* x    = (const float*)d_in[0];
  const int*   ei   = (const int*)d_in[1];
  const float* Wa   = (const float*)d_in[2];
  const float* ba   = (const float*)d_in[3];
  const float* Wp   = (const float*)d_in[4];
  const float* bp   = (const float*)d_in[5];
  const float* e1W1 = (const float*)d_in[6];
  const float* e1b1 = (const float*)d_in[7];
  const float* e1W2 = (const float*)d_in[8];
  const float* e1b2 = (const float*)d_in[9];
  const float* e1lW = (const float*)d_in[10];
  const float* e1lb = (const float*)d_in[11];
  const float* e2W1 = (const float*)d_in[12];
  const float* e2b1 = (const float*)d_in[13];
  const float* e2W2 = (const float*)d_in[14];
  const float* e2b2 = (const float*)d_in[15];
  const float* e2lW = (const float*)d_in[16];
  const float* e2lb = (const float*)d_in[17];
  const float* Wout = (const float*)d_in[18];
  const float* bout = (const float*)d_in[19];

  float* out = (float*)d_out;
  float* w   = (float*)d_ws;
  float* Sp   = w + SP_OFF;
  float* deg  = w + DEG_OFF;
  float* ZG   = w + ZG_OFF;
  float* SS   = w + SS_OFF;
  float* ADJ  = w + ADJ_OFF;
  float* VOL  = w + VOL_OFF;
  float* EA   = w + EA_OFF;
  float* EB   = w + EB_OFF;
  float* EM   = w + EM_OFF;
  float* EEMB = w + EEMB_OFF;
  float* H1   = w + H1_OFF;
  float* H2   = w + H2_OFF;
  unsigned short* wpb = (unsigned short*)(w + WPB_OFF);
  unsigned short* wab = (unsigned short*)(w + WAB_OFF);
  unsigned short* xbf = (unsigned short*)(w + XBF_OFF);

  k_convx<<<(NN * (DD / 8) + 255) / 256, 256, 0, stream>>>(x, xbf);
  k_prep<<<1024, 256, 0, stream>>>(Wp, Wa, deg, ZG, SS, ADJ, VOL, wpb, wab);
  k_s<<<512, 128, 0, stream>>>(xbf, wab, ba, Sp, out + O_S);
  k_edges<<<256, 256, 0, stream>>>(ei, Sp, ADJ, deg);      // 2048 waves
  k_reduce<<<400, 256, 0, stream>>>(Sp, deg, SS, VOL);
  k_loss<<<1, 32, 0, stream>>>(SS, ADJ, VOL, out + O_ML, out + O_OL);
  k_xpz<<<512, 256, 0, stream>>>(xbf, wpb, bp, Sp, ZG);    // 8 col-groups x 64 node splits
  k_copyz<<<(KK * DD + 255) / 256, 256, 0, stream>>>(ZG, out + O_Z);

  // explainer layer 1 (input: Z rows 0..29 of ZG)
  k_eab <<<(2 * KK * HH + 255) / 256, 256, 0, stream>>>(ZG, e1W1, EA, EB, DD, DD);
  k_elog<<<(KK * KK * 32 + 255) / 256, 256, 0, stream>>>(EA, EB, e1b1, e1W2, e1b2, ADJ, EM, DD);
  k_eemb<<<(KK * DD + 255) / 256, 256, 0, stream>>>(ZG, EM, ADJ, EEMB, DD);
  k_elin<<<(KK * HH + 255) / 256, 256, 0, stream>>>(EEMB, e1lW, e1lb, H1, DD, HH, 1);
  // explainer layer 2 (input: h1)
  k_eab <<<(2 * KK * HH + 255) / 256, 256, 0, stream>>>(H1, e2W1, EA, EB, HH, HH);
  k_elog<<<(KK * KK * 32 + 255) / 256, 256, 0, stream>>>(EA, EB, e2b1, e2W2, e2b2, ADJ, EM, HH);
  k_eemb<<<(KK * HH + 255) / 256, 256, 0, stream>>>(H1, EM, ADJ, EEMB, HH);
  k_elin<<<(KK * HH + 255) / 256, 256, 0, stream>>>(EEMB, e2lW, e2lb, H2, HH, HH, 1);
  // final linear -> out [30,128]
  k_elin<<<(KK * OO + 255) / 256, 256, 0, stream>>>(H2, Wout, bout, out + O_OUT, HH, OO, 0);
}
